// MultiProductAttention_66872640799276
// MI455X (gfx1250) — matrix operand order repaired and checked by reading, not compiled, not run on hardware
//
#include <hip/hip_runtime.h>
#include <math.h>

// Problem constants (match reference)
#define BB   2
#define NN   2048
#define DD   1024
#define HH   16
#define HDIM 64
#define MTOT (BB * NN)          // 4096 rows
#define OUTD (HH * HDIM)        // 1024

typedef _Float16 v8h  __attribute__((ext_vector_type(8)));
typedef _Float16 v16h __attribute__((ext_vector_type(16)));
typedef float    v8f  __attribute__((ext_vector_type(8)));

// ---------------------------------------------------------------------------
// WMMA operand loaders (wave32, 16x16x32 f16 shapes)
// A operand (16x32, M=lane&15): halves e -> k = k0 + 16*g + e
// ---------------------------------------------------------------------------
static __device__ inline v16h load_opA(const _Float16* row, int k0, int g) {
  const v8h t0 = *(const v8h*)(row + k0 + 16 * g);
  const v8h t1 = *(const v8h*)(row + k0 + 16 * g + 8);
  v16h r;
#pragma unroll
  for (int i = 0; i < 8; ++i) { r[i] = t0[i]; r[8 + i] = t1[i]; }
  return r;
}

// B operand (32x16, N=lane&15): halves e -> k = k0 + 16*g + e
static __device__ inline v16h load_opB(const _Float16* row, int k0, int g) {
  const v8h t0 = *(const v8h*)(row + k0 + 16 * g);
  const v8h t1 = *(const v8h*)(row + k0 + 16 * g + 8);
  v16h r;
#pragma unroll
  for (int i = 0; i < 8; ++i) { r[i] = t0[i]; r[8 + i] = t1[i]; }
  return r;
}

// ---------------------------------------------------------------------------
// Big WMMA GEMM with async global->LDS double buffering.
// Block = 256 threads (8 waves). Block tile 128(M) x 64(N), K step 32.
// ---------------------------------------------------------------------------
#define MBLK 128
#define NBLK 64
#define KSTEP 32

__global__ __launch_bounds__(256)
void wmma_gemm_f16_lds(const _Float16* __restrict__ A, const _Float16* __restrict__ Bt,
                       float* __restrict__ C, int M, int N, int K) {
  __shared__ __align__(16) _Float16 Abuf[2][MBLK][KSTEP];
  __shared__ __align__(16) _Float16 Bbuf[2][NBLK][KSTEP];

  const int t    = threadIdx.x;
  const int lane = t & 31;
  const int w    = t >> 5;
  const int l15  = lane & 15;
  const int g    = lane >> 4;
  const int wm   = w & 3;
  const int wn   = w >> 2;
  const int m0   = blockIdx.y * MBLK;
  const int n0   = blockIdx.x * NBLK;

  auto issue = [&](int s, int buf) {
    const int k0 = s * KSTEP;
#pragma unroll
    for (int c = 0; c < 2; ++c) {           // A slab: 128 rows x 64B
      int idx = t + 256 * c;
      int row = idx >> 2, cic = idx & 3;
      const _Float16* src = A + (size_t)(m0 + row) * K + k0 + cic * 8;
      unsigned dst = (unsigned)(uintptr_t)&Abuf[buf][row][cic * 8];
      asm volatile("global_load_async_to_lds_b128 %0, %1, off"
                   :: "v"(dst), "v"(src) : "memory");
    }
    {                                        // B slab: 64 rows x 64B
      int row = t >> 2, cic = t & 3;
      const _Float16* src = Bt + (size_t)(n0 + row) * K + k0 + cic * 8;
      unsigned dst = (unsigned)(uintptr_t)&Bbuf[buf][row][cic * 8];
      asm volatile("global_load_async_to_lds_b128 %0, %1, off"
                   :: "v"(dst), "v"(src) : "memory");
    }
  };

  v8f acc[2][2];
#pragma unroll
  for (int mt = 0; mt < 2; ++mt)
#pragma unroll
    for (int nt = 0; nt < 2; ++nt)
#pragma unroll
      for (int j = 0; j < 8; ++j) acc[mt][nt][j] = 0.0f;

  const int S = K / KSTEP;
  issue(0, 0);

  for (int s = 0; s < S; ++s) {
    const int buf = s & 1;
    if (s + 1 < S) {
      issue(s + 1, buf ^ 1);
      asm volatile("s_wait_asynccnt 0x3" ::: "memory");
    } else {
      asm volatile("s_wait_asynccnt 0x0" ::: "memory");
    }
    __syncthreads();

    const _Float16* arow0 = &Abuf[buf][wm * 32 + l15][0];
    const _Float16* arow1 = &Abuf[buf][wm * 32 + 16 + l15][0];
    const _Float16* brow0 = &Bbuf[buf][wn * 32 + l15][0];
    const _Float16* brow1 = &Bbuf[buf][wn * 32 + 16 + l15][0];
    const v16h a0 = load_opA(arow0, 0, g);
    const v16h a1 = load_opA(arow1, 0, g);
    const v16h b0 = load_opB(brow0, 0, g);
    const v16h b1 = load_opB(brow1, 0, g);
    acc[0][0] = __builtin_amdgcn_wmma_f32_16x16x32_f16(false, a0, false, b0, (short)0, acc[0][0], false, false);
    acc[0][1] = __builtin_amdgcn_wmma_f32_16x16x32_f16(false, a0, false, b1, (short)0, acc[0][1], false, false);
    acc[1][0] = __builtin_amdgcn_wmma_f32_16x16x32_f16(false, a1, false, b0, (short)0, acc[1][0], false, false);
    acc[1][1] = __builtin_amdgcn_wmma_f32_16x16x32_f16(false, a1, false, b1, (short)0, acc[1][1], false, false);
    __syncthreads();
  }

#pragma unroll
  for (int mt = 0; mt < 2; ++mt)
#pragma unroll
    for (int nt = 0; nt < 2; ++nt)
#pragma unroll
      for (int j = 0; j < 8; ++j) {
        int mr = m0 + wm * 32 + mt * 16 + j + 8 * g;
        int nc = n0 + wn * 32 + nt * 16 + l15;
        C[(size_t)mr * N + nc] = acc[mt][nt][j];
      }
}

// ---------------------------------------------------------------------------
// Small WMMA GEMM (N < 64): one wave per 16x16 tile, direct global loads.
// ---------------------------------------------------------------------------
__global__ __launch_bounds__(32)
void wmma_gemm_f16(const _Float16* __restrict__ A, const _Float16* __restrict__ Bt,
                   float* __restrict__ C, int M, int N, int K) {
  const int lane = threadIdx.x;
  const int l15  = lane & 15;
  const int g    = lane >> 4;
  const int n0   = blockIdx.x * 16;
  const int m0   = blockIdx.y * 16;

  const _Float16* arow = A  + (size_t)(m0 + l15) * K;
  const _Float16* brow = Bt + (size_t)(n0 + l15) * K;

  v8f acc;
#pragma unroll
  for (int j = 0; j < 8; ++j) acc[j] = 0.0f;

  for (int k0 = 0; k0 < K; k0 += 32) {
    if (k0 + 64 <= K) {
      __builtin_prefetch(arow + k0 + 32, 0, 3);
      __builtin_prefetch(brow + k0 + 32, 0, 3);
    }
    const v16h a = load_opA(arow, k0, g);
    const v16h b = load_opB(brow, k0, g);
    acc = __builtin_amdgcn_wmma_f32_16x16x32_f16(false, a, false, b,
                                                 (short)0, acc, false, false);
  }
#pragma unroll
  for (int j = 0; j < 8; ++j)
    C[(size_t)(m0 + j + 8 * g) * N + n0 + l15] = acc[j];
}

// ---------------------------------------------------------------------------
// Elementwise helpers
// ---------------------------------------------------------------------------
__global__ __launch_bounds__(256)
void cvt_f32_to_f16(const float* __restrict__ src, _Float16* __restrict__ dst, int n) {
  int i = blockIdx.x * blockDim.x + threadIdx.x;
  if (i < n) dst[i] = (_Float16)src[i];
}

__global__ __launch_bounds__(256)
void transpose_to_f16(const float* __restrict__ W, _Float16* __restrict__ Wt, int K, int N) {
  int i = blockIdx.x * blockDim.x + threadIdx.x;
  if (i >= K * N) return;
  int k = i / N, n = i % N;
  Wt[(size_t)n * K + k] = (_Float16)W[i];
}

__global__ __launch_bounds__(256)
void sigmoid_inplace(float* __restrict__ buf, int n) {
  int i = blockIdx.x * blockDim.x + threadIdx.x;
  if (i < n) buf[i] = 1.0f / (1.0f + __expf(-buf[i]));
}

// RoPE on q + relayout (b,n,h*64) f32 -> (b,h,n,64) f16
__global__ __launch_bounds__(256)
void rope_q_kernel(const float* __restrict__ q_tmp, _Float16* __restrict__ qbuf) {
  int i = blockIdx.x * blockDim.x + threadIdx.x;
  if (i >= BB * NN * HH * 32) return;
  int j = i & 31;  int t = i >> 5;
  int h = t & 15;  t >>= 4;
  int n = t % NN;  int b = t / NN;
  float inv_freq = __expf(-(float)j * (9.210340371976184f / 32.0f));
  float ang = (float)n * inv_freq;
  float c = cosf(ang), s = sinf(ang);
  size_t base = ((size_t)(b * NN + n)) * OUTD + h * HDIM;
  float x1 = q_tmp[base + j];
  float x2 = q_tmp[base + 32 + j];
  size_t ob = (((size_t)(b * HH + h)) * NN + n) * HDIM;
  qbuf[ob + j]      = (_Float16)(x1 * c - x2 * s);
  qbuf[ob + 32 + j] = (_Float16)(x2 * c + x1 * s);
}

// kv epilogue: RoPE on k -> kbuf (b,n,64) f16; v -> vtbuf (b,64,n) f16 (transposed)
__global__ __launch_bounds__(256)
void kv_epilogue(const float* __restrict__ kv_tmp,
                 _Float16* __restrict__ kbuf, _Float16* __restrict__ vtbuf) {
  int i = blockIdx.x * blockDim.x + threadIdx.x;
  if (i >= BB * NN * 32) return;
  int j = i & 31;  int t = i >> 5;
  int n = t % NN;  int b = t / NN;
  size_t base = ((size_t)(b * NN + n)) * (2 * HDIM);
  float inv_freq = __expf(-(float)j * (9.210340371976184f / 32.0f));
  float ang = (float)n * inv_freq;
  float c = cosf(ang), s = sinf(ang);
  float k1 = kv_tmp[base + j];
  float k2 = kv_tmp[base + 32 + j];
  size_t kb = ((size_t)(b * NN + n)) * HDIM;
  kbuf[kb + j]      = (_Float16)(k1 * c - k2 * s);
  kbuf[kb + 32 + j] = (_Float16)(k2 * c + k1 * s);
  float v1 = kv_tmp[base + HDIM + j];
  float v2 = kv_tmp[base + HDIM + 32 + j];
  vtbuf[((size_t)(b * HDIM + j)) * NN + n]      = (_Float16)v1;
  vtbuf[((size_t)(b * HDIM + 32 + j)) * NN + n] = (_Float16)v2;
}

// ---------------------------------------------------------------------------
// Flash attention with async-LDS staged K/V shared by 8 waves.
// Block = 256 threads: wave w handles query tile q0 = (blockIdx.x*8+w)*16 of
// head (blockIdx.y), batch (blockIdx.z). The 32-key K slab (32x64 f16) and
// V^T slab (64x32 f16) are DMA'd to LDS (double buffered) and reused by all
// 8 waves: 8x less L2 traffic, DMA of chunk n+1 overlaps compute of chunk n.
// Causally-finished waves skip compute but keep the barrier schedule so EXEC
// is all-ones around every WMMA.
// ---------------------------------------------------------------------------
__global__ __launch_bounds__(256)
void flash_attn_lds_kernel(const _Float16* __restrict__ qbuf,
                           const _Float16* __restrict__ kbuf,
                           const _Float16* __restrict__ vtbuf,
                           const float* __restrict__ headbuf,
                           _Float16* __restrict__ aobuf) {
  __shared__ __align__(16) _Float16 Kb[2][32][HDIM];   // keys x kdims
  __shared__ __align__(16) _Float16 Vb[2][HDIM][32];   // dims x keys

  const int t    = threadIdx.x;
  const int lane = t & 31;
  const int w    = t >> 5;
  const int l15  = lane & 15;
  const int g    = lane >> 4;
  const int h    = blockIdx.y;
  const int b    = blockIdx.z;
  const int q0   = (blockIdx.x * 8 + w) * 16;
  const int qcol = q0 + l15;

  auto issue = [&](int ch, int buf) {
    const int k0 = ch * 32;
    {  // K slab: 32 rows x 128B = 256 x 16B chunks
      int row = t >> 3, cic = t & 7;
      const _Float16* src = kbuf + ((size_t)b * NN + k0 + row) * HDIM + cic * 8;
      unsigned dst = (unsigned)(uintptr_t)&Kb[buf][row][cic * 8];
      asm volatile("global_load_async_to_lds_b128 %0, %1, off"
                   :: "v"(dst), "v"(src) : "memory");
    }
    {  // V^T slab: 64 rows x 64B = 256 x 16B chunks
      int row = t >> 2, cic = t & 3;
      const _Float16* src = vtbuf + ((size_t)b * HDIM + row) * NN + k0 + cic * 8;
      unsigned dst = (unsigned)(uintptr_t)&Vb[buf][row][cic * 8];
      asm volatile("global_load_async_to_lds_b128 %0, %1, off"
                   :: "v"(dst), "v"(src) : "memory");
    }
  };

  // Per-wave Q^T operands (registers, reused across all key chunks).
  const _Float16* qrow = qbuf + (((size_t)(b * HH + h)) * NN + qcol) * HDIM;
  v16h qB0 = load_opB(qrow, 0, g);
  v16h qB1 = load_opB(qrow, 32, g);

  v8f o[4];
#pragma unroll
  for (int dt = 0; dt < 4; ++dt)
#pragma unroll
    for (int j = 0; j < 8; ++j) o[dt][j] = 0.0f;

  float run_m = -3.0e38f, run_l = 0.0f;
  const int nch_w   = (q0 + 15) / 32 + 1;     // this wave's causal chunk count
  const int nch_blk = blockIdx.x * 4 + 4;     // block max (last wave's count)

  issue(0, 0);
  for (int ch = 0; ch < nch_blk; ++ch) {
    const int buf = ch & 1;
    if (ch + 1 < nch_blk) {
      issue(ch + 1, buf ^ 1);
      asm volatile("s_wait_asynccnt 0x2" ::: "memory");  // stage = 2 async ops
    } else {
      asm volatile("s_wait_asynccnt 0x0" ::: "memory");
    }
    __syncthreads();

    if (ch < nch_w) {
      const int k0 = ch * 32;
      float s[2][8];
#pragma unroll
      for (int kt = 0; kt < 2; ++kt) {
        const _Float16* krow = &Kb[buf][16 * kt + l15][0];
        v8f st;
#pragma unroll
        for (int j = 0; j < 8; ++j) st[j] = 0.0f;
        v16h ak = load_opA(krow, 0, g);
        st = __builtin_amdgcn_wmma_f32_16x16x32_f16(false, ak, false, qB0,
                                                    (short)0, st, false, false);
        ak = load_opA(krow, 32, g);
        st = __builtin_amdgcn_wmma_f32_16x16x32_f16(false, ak, false, qB1,
                                                    (short)0, st, false, false);
#pragma unroll
        for (int j = 0; j < 8; ++j) {
          int key = k0 + 16 * kt + j + 8 * g;
          float kh = headbuf[((size_t)b * NN + key) * (2 * HH) + h];
          float val = st[j] * kh * 0.125f;  // 1/sqrt(64)
          s[kt][j] = (key > qcol) ? -3.0e38f : val;
        }
      }
      float cm = s[0][0];
#pragma unroll
      for (int kt = 0; kt < 2; ++kt)
#pragma unroll
        for (int j = 0; j < 8; ++j) cm = fmaxf(cm, s[kt][j]);
      cm = fmaxf(cm, __shfl_xor(cm, 16));
      float new_m = fmaxf(run_m, cm);
      float alpha = __expf(run_m - new_m);

      float csum = 0.0f;
      float pv[2][8];
#pragma unroll
      for (int kt = 0; kt < 2; ++kt)
#pragma unroll
        for (int j = 0; j < 8; ++j) {
          int key = k0 + 16 * kt + j + 8 * g;
          float p = (s[kt][j] > -1.0e37f) ? __expf(s[kt][j] - new_m) : 0.0f;
          csum += p;  // denominator excludes v_head
          float vh = headbuf[((size_t)b * NN + key) * (2 * HH) + HH + h];
          pv[kt][j] = p * vh;
        }
      csum += __shfl_xor(csum, 16);
      run_l = run_l * alpha + csum;
      run_m = new_m;

      float ov0[8], ov1[8];
#pragma unroll
      for (int j = 0; j < 8; ++j) {
        ov0[j] = __shfl_xor(pv[0][j], 16);
        ov1[j] = __shfl_xor(pv[1][j], 16);
      }
      v16h pt;
      if (g == 0) {
#pragma unroll
        for (int e = 0; e < 8; ++e) { pt[e] = (_Float16)pv[0][e]; pt[8 + e] = (_Float16)ov0[e]; }
      } else {
#pragma unroll
        for (int e = 0; e < 8; ++e) { pt[e] = (_Float16)ov1[e]; pt[8 + e] = (_Float16)pv[1][e]; }
      }

#pragma unroll
      for (int dt = 0; dt < 4; ++dt) {
        const _Float16* vrow = &Vb[buf][dt * 16 + l15][0];
        v16h av = load_opA(vrow, 0, g);
#pragma unroll
        for (int j = 0; j < 8; ++j) o[dt][j] *= alpha;
        o[dt] = __builtin_amdgcn_wmma_f32_16x16x32_f16(false, av, false, pt,
                                                       (short)0, o[dt], false, false);
      }
    }
    __syncthreads();
  }

  const float inv = 1.0f / run_l;
#pragma unroll
  for (int dt = 0; dt < 4; ++dt)
#pragma unroll
    for (int j = 0; j < 8; ++j) {
      int d = dt * 16 + j + 8 * g;
      aobuf[((size_t)(b * NN + qcol)) * OUTD + h * HDIM + d] = (_Float16)(o[dt][j] * inv);
    }
}

// ---------------------------------------------------------------------------
// Host launcher
// ---------------------------------------------------------------------------
extern "C" void kernel_launch(void* const* d_in, const int* in_sizes, int n_in,
                              void* d_out, int out_size, void* d_ws, size_t ws_size,
                              hipStream_t stream) {
  (void)in_sizes; (void)n_in; (void)out_size; (void)ws_size;
  const float* x    = (const float*)d_in[0];
  const float* Wq   = (const float*)d_in[1];
  const float* Wkv  = (const float*)d_in[2];
  const float* Wkvh = (const float*)d_in[3];
  const float* Wo   = (const float*)d_in[4];
  float* out = (float*)d_out;

  char*  ws  = (char*)d_ws;
  size_t off = 0;
  auto take = [&](size_t bytes) -> void* {
    void* p = ws + off;
    off += (bytes + 255) & ~(size_t)255;
    return p;
  };
  _Float16* xh      = (_Float16*)take((size_t)MTOT * DD * 2);
  _Float16* WqT     = (_Float16*)take((size_t)DD * OUTD * 2);
  _Float16* WkvT    = (_Float16*)take((size_t)DD * 2 * HDIM * 2);
  _Float16* WkvhT   = (_Float16*)take((size_t)DD * 2 * HH * 2);
  _Float16* WoT     = (_Float16*)take((size_t)OUTD * DD * 2);
  float*    q_tmp   = (float*)take((size_t)MTOT * OUTD * 4);
  float*    kv_tmp  = (float*)take((size_t)MTOT * 2 * HDIM * 4);
  float*    headbuf = (float*)take((size_t)MTOT * 2 * HH * 4);
  _Float16* qbuf    = (_Float16*)take((size_t)BB * HH * NN * HDIM * 2);
  _Float16* kbuf    = (_Float16*)take((size_t)BB * NN * HDIM * 2);
  _Float16* vtbuf   = (_Float16*)take((size_t)BB * HDIM * NN * 2);
  _Float16* aobuf   = (_Float16*)take((size_t)MTOT * OUTD * 2);

  // 1) Downconvert activations + transpose/downconvert weights.
  {
    int n = MTOT * DD;
    cvt_f32_to_f16<<<(n + 255) / 256, 256, 0, stream>>>(x, xh, n);
  }
  transpose_to_f16<<<(DD * OUTD + 255) / 256, 256, 0, stream>>>(Wq, WqT, DD, OUTD);
  transpose_to_f16<<<(DD * 2 * HDIM + 255) / 256, 256, 0, stream>>>(Wkv, WkvT, DD, 2 * HDIM);
  transpose_to_f16<<<(DD * 2 * HH + 255) / 256, 256, 0, stream>>>(Wkvh, WkvhT, DD, 2 * HH);
  transpose_to_f16<<<(OUTD * DD + 255) / 256, 256, 0, stream>>>(Wo, WoT, OUTD, DD);

  // 2) Projections via WMMA GEMM (LDS double-buffered for the big ones).
  wmma_gemm_f16_lds<<<dim3(OUTD / NBLK, MTOT / MBLK), 256, 0, stream>>>(xh, WqT, q_tmp, MTOT, OUTD, DD);
  wmma_gemm_f16_lds<<<dim3((2 * HDIM) / NBLK, MTOT / MBLK), 256, 0, stream>>>(xh, WkvT, kv_tmp, MTOT, 2 * HDIM, DD);
  wmma_gemm_f16<<<dim3((2 * HH) / 16, MTOT / 16), 32, 0, stream>>>(xh, WkvhT, headbuf, MTOT, 2 * HH, DD);

  // 3) Epilogues: sigmoid gates, RoPE + relayout.
  sigmoid_inplace<<<(MTOT * 2 * HH + 255) / 256, 256, 0, stream>>>(headbuf, MTOT * 2 * HH);
  rope_q_kernel<<<(BB * NN * HH * 32 + 255) / 256, 256, 0, stream>>>(q_tmp, qbuf);
  kv_epilogue<<<(BB * NN * 32 + 255) / 256, 256, 0, stream>>>(kv_tmp, kbuf, vtbuf);

  // 4) Flash attention: async-LDS staged shared K/V, 8 q-tiles per block.
  flash_attn_lds_kernel<<<dim3(NN / 128, HH, BB), 256, 0, stream>>>(qbuf, kbuf, vtbuf, headbuf, aobuf);

  // 5) Output projection -> fp32 d_out.
  wmma_gemm_f16_lds<<<dim3(DD / NBLK, MTOT / MBLK), 256, 0, stream>>>(aobuf, WoT, out, MTOT, DD, OUTD);
}
